// LSTMDecoder_77060303225563
// MI455X (gfx1250) — matrix operand order repaired and checked by reading, not compiled, not run on hardware
//
#include <hip/hip_runtime.h>
#include <stdint.h>

#define BB 128
#define LL 1024
#define II 256
#define OO 256
#define HH 1024

typedef __attribute__((ext_vector_type(16))) __bf16 v16bf;
typedef __attribute__((ext_vector_type(8)))  float  v8f;

__device__ __forceinline__ unsigned short f2bf(float f) {
  uint32_t u = __float_as_uint(f);
  return (unsigned short)((u + 0x7FFFu + ((u >> 16) & 1u)) >> 16);  // RNE
}
__device__ __forceinline__ float sigmoidf_(float x) { return 1.0f / (1.0f + __expf(-x)); }
__device__ __forceinline__ float tanhf_(float x)    { float e = __expf(2.0f * x); return 1.0f - 2.0f / (e + 1.0f); }

// ==========================================================================
// WMMA fragment layouts (bf16, 16x16x32, wave32):
//  A (16x32, MxK): lane l holds M = l&15;  k_local = (e/8)*16 + (l/16)*8 + (e%8)
//  B (32x16, KxN): lane l holds N = l&15;  k_local = (e/8)*16 + (l/16)*8 + (e%8)
//  C/D (16x16):    lane l, vgpr r -> M = r + 8*(l/16), N = l&15
//  Fragments stored as [block..][lane(32)][e(16)] -> contiguous 32B per lane.
// ==========================================================================

// ---- W [Nrows x Kdim] fp32 row-major -> B-fragment bf16 panels [nb][kk][lane][16]
__global__ void swizzle_b_kernel(const float* __restrict__ W, uint16_t* __restrict__ out,
                                 int Kdim, int total) {
  int idx = blockIdx.x * blockDim.x + threadIdx.x;
  if (idx >= total) return;
  int numK = Kdim >> 5;
  int e = idx & 15;
  int t = idx >> 4;
  int l = t & 31; t >>= 5;
  int kk = t % numK;
  int nb = t / numK;
  int n = (nb << 4) + (l & 15);
  int k = (kk << 5) + ((e >> 3) << 4) + ((l >> 4) << 3) + (e & 7);
  out[idx] = f2bf(W[(size_t)n * Kdim + k]);
}

// ---- trg [B,L,I] fp32 -> A-fragment bf16 panels [t][mb(8)][kk(8)][lane][16]
__global__ void swizzle_x_kernel(const float* __restrict__ trg, uint16_t* __restrict__ out,
                                 int total) {
  int idx = blockIdx.x * blockDim.x + threadIdx.x;
  if (idx >= total) return;
  int e  = idx & 15;
  int l  = (idx >> 4) & 31;
  int kk = (idx >> 9) & 7;    // I/32  = 8
  int mb = (idx >> 12) & 7;   // B/16  = 8
  int t  = idx >> 15;
  int m  = (mb << 4) + (l & 15);
  int k  = (kk << 5) + ((e >> 3) << 4) + ((l >> 4) << 3) + (e & 7);
  out[idx] = f2bf(trg[((size_t)m * LL + t) * II + k]);
}

__global__ void bias_kernel(const float* __restrict__ a, const float* __restrict__ b,
                            float* __restrict__ o, int n) {
  int i = blockIdx.x * blockDim.x + threadIdx.x;
  if (i < n) o[i] = a[i] + b[i];
}

#define FRAG_BYTES 1024   // one 16x16x32 fragment = 32 lanes * 32 B

__device__ __forceinline__ v16bf ldfrag(const char* p, int byteOff) {
  return *(const v16bf*)(p + byteOff);
}

#define WMMA_BF16(A, Bm, C) \
  __builtin_amdgcn_wmma_f32_16x16x32_bf16(false, (A), false, (Bm), (short)0, (C), false, false)

// ==========================================================================
// One LSTM timestep + fused output projection of the PREVIOUS step's h.
// grid = 64 (H col blocks), block = 256 (8 waves; wave = batch row block).
// WGs with j<16 additionally compute out[:, emit_t, :] cols [16j,16j+16)
// on the same h_in A-fragment stream (5th B-stream + 5th accumulator).
// h kept in A-fragment layout [mb(8)][kk(32)][lane][16], double buffered.
// waves_per_eu(2): real occupancy is 2 waves/SIMD; free the register file so
// the scheduler hoists loads for latency hiding instead of minimizing regs.
// ==========================================================================
__global__ void __launch_bounds__(256) __attribute__((amdgpu_waves_per_eu(2)))
lstm_step_kernel(const uint16_t* __restrict__ xsw, int t_in,
                 const uint16_t* __restrict__ wih,
                 const uint16_t* __restrict__ whh,
                 const float* __restrict__ bias4,
                 const uint16_t* __restrict__ h_in,
                 uint16_t* __restrict__ h_out,
                 float* __restrict__ cst,
                 const uint16_t* __restrict__ wout,
                 const float* __restrict__ b_out,
                 float* __restrict__ outbuf, int emit_t) {
  const int lane = threadIdx.x & 31;
  const int wave = threadIdx.x >> 5;      // batch row block 0..7
  const int j    = blockIdx.x;            // H column block 0..63
  const size_t laneB = (size_t)lane * 32; // lane byte offset inside fragment

  v8f acc0 = {}, acc1 = {}, acc2 = {}, acc3 = {}, acc4 = {};
  const bool do_out = (j < (OO / 16)) && (emit_t >= 0);   // workgroup-uniform

  // ---- input projection: K = II = 256 (8 chunks) ----
  {
    const char* pa  = (const char*)xsw + (size_t)((t_in << 3) + wave) * 8 * FRAG_BYTES + laneB;
    const char* pb0 = (const char*)wih + (size_t)(0 * 64 + j) * 8 * FRAG_BYTES + laneB;
    const char* pb1 = (const char*)wih + (size_t)(1 * 64 + j) * 8 * FRAG_BYTES + laneB;
    const char* pb2 = (const char*)wih + (size_t)(2 * 64 + j) * 8 * FRAG_BYTES + laneB;
    const char* pb3 = (const char*)wih + (size_t)(3 * 64 + j) * 8 * FRAG_BYTES + laneB;
    #pragma unroll
    for (int kk = 0; kk < (II >> 5); ++kk) {
      v16bf a  = ldfrag(pa,  kk * FRAG_BYTES);
      v16bf b0 = ldfrag(pb0, kk * FRAG_BYTES);
      v16bf b1 = ldfrag(pb1, kk * FRAG_BYTES);
      v16bf b2 = ldfrag(pb2, kk * FRAG_BYTES);
      v16bf b3 = ldfrag(pb3, kk * FRAG_BYTES);
      acc0 = WMMA_BF16(a, b0, acc0);
      acc1 = WMMA_BF16(a, b1, acc1);
      acc2 = WMMA_BF16(a, b2, acc2);
      acc3 = WMMA_BF16(a, b3, acc3);
    }
  }

  // ---- recurrent projection (K = HH = 1024, 32 chunks), + fused out-proj ----
  {
    const char* pa  = (const char*)h_in + (size_t)wave * 32 * FRAG_BYTES + laneB;
    const char* pb0 = (const char*)whh + (size_t)(0 * 64 + j) * 32 * FRAG_BYTES + laneB;
    const char* pb1 = (const char*)whh + (size_t)(1 * 64 + j) * 32 * FRAG_BYTES + laneB;
    const char* pb2 = (const char*)whh + (size_t)(2 * 64 + j) * 32 * FRAG_BYTES + laneB;
    const char* pb3 = (const char*)whh + (size_t)(3 * 64 + j) * 32 * FRAG_BYTES + laneB;
    if (do_out) {
      const char* pb4 = (const char*)wout + (size_t)j * 32 * FRAG_BYTES + laneB;
      #pragma unroll
      for (int kk = 0; kk < (HH >> 5); ++kk) {
        v16bf a  = ldfrag(pa,  kk * FRAG_BYTES);
        v16bf b0 = ldfrag(pb0, kk * FRAG_BYTES);
        v16bf b1 = ldfrag(pb1, kk * FRAG_BYTES);
        v16bf b2 = ldfrag(pb2, kk * FRAG_BYTES);
        v16bf b3 = ldfrag(pb3, kk * FRAG_BYTES);
        v16bf b4 = ldfrag(pb4, kk * FRAG_BYTES);
        acc0 = WMMA_BF16(a, b0, acc0);
        acc1 = WMMA_BF16(a, b1, acc1);
        acc2 = WMMA_BF16(a, b2, acc2);
        acc3 = WMMA_BF16(a, b3, acc3);
        acc4 = WMMA_BF16(a, b4, acc4);
      }
    } else {
      #pragma unroll
      for (int kk = 0; kk < (HH >> 5); ++kk) {
        v16bf a  = ldfrag(pa,  kk * FRAG_BYTES);
        v16bf b0 = ldfrag(pb0, kk * FRAG_BYTES);
        v16bf b1 = ldfrag(pb1, kk * FRAG_BYTES);
        v16bf b2 = ldfrag(pb2, kk * FRAG_BYTES);
        v16bf b3 = ldfrag(pb3, kk * FRAG_BYTES);
        acc0 = WMMA_BF16(a, b0, acc0);
        acc1 = WMMA_BF16(a, b1, acc1);
        acc2 = WMMA_BF16(a, b2, acc2);
        acc3 = WMMA_BF16(a, b3, acc3);
      }
    }
  }

  // ---- gates + state update ----
  const int n   = lane & 15;
  const int hi  = lane >> 4;
  const int col = (j << 4) + n;
  float bi  = bias4[0 * HH + col];
  float bfv = bias4[1 * HH + col];
  float bg  = bias4[2 * HH + col];
  float bo  = bias4[3 * HH + col];

  // h_out swizzled-store coords (A layout for next step / out proj):
  //   m = wave*16 + hi*8 + r ; k = col ; mb = wave ; kk_h = col/32
  //   lane_t = (hi*8 + r) + 16*((n/8)&1) ; e = (j&1)*8 + (n&7)
  const int kk_h   = j >> 1;
  const int eh     = ((j & 1) << 3) + (n & 7);
  const int ltbase = (hi << 3) + ((n >> 3) << 4);
  uint16_t* hob = h_out + ((((size_t)wave << 5) + kk_h) << 5) * 16;

  #pragma unroll
  for (int r = 0; r < 8; ++r) {
    int mg = (wave << 4) + (hi << 3) + r;
    float zi = acc0[r] + bi;
    float zf = acc1[r] + bfv;
    float zg = acc2[r] + bg;
    float zo = acc3[r] + bo;
    size_t off = (size_t)mg * HH + col;
    float cv = cst[off];
    float cn = sigmoidf_(zf) * cv + sigmoidf_(zi) * tanhf_(zg);
    float hn = sigmoidf_(zo) * tanhf_(cn);
    cst[off] = cn;
    hob[(size_t)(ltbase + r) * 16 + eh] = f2bf(hn);
  }

  // ---- fused output store: out[:, emit_t, cols 16j..16j+16) ----
  if (do_out) {
    float bb = b_out[col];
    #pragma unroll
    for (int r = 0; r < 8; ++r) {
      int mg = (wave << 4) + (hi << 3) + r;
      outbuf[((size_t)mg * LL + emit_t) * OO + col] = acc4[r] + bb;
    }
  }
}

// ==========================================================================
// Standalone out projection (used once, for the final step's h).
// ==========================================================================
__global__ void __launch_bounds__(256) __attribute__((amdgpu_waves_per_eu(2)))
out_proj_kernel(const uint16_t* __restrict__ hsw, const uint16_t* __restrict__ wout,
                const float* __restrict__ b_out, float* __restrict__ out, int t) {
  const int lane = threadIdx.x & 31;
  const int wave = threadIdx.x >> 5;
  const int j    = blockIdx.x;            // O column block 0..15
  const size_t laneB = (size_t)lane * 32;

  const char* pa = (const char*)hsw  + (size_t)wave * 32 * FRAG_BYTES + laneB;
  const char* pb = (const char*)wout + (size_t)j * 32 * FRAG_BYTES + laneB;

  v8f acc = {};
  #pragma unroll
  for (int kk = 0; kk < (HH >> 5); ++kk) {
    v16bf a = ldfrag(pa, kk * FRAG_BYTES);
    v16bf b = ldfrag(pb, kk * FRAG_BYTES);
    acc = WMMA_BF16(a, b, acc);
  }
  float bb = b_out[(j << 4) + (lane & 15)];
  #pragma unroll
  for (int r = 0; r < 8; ++r) {
    int mg = (wave << 4) + ((lane >> 4) << 3) + r;
    out[((size_t)mg * LL + t) * OO + (j << 4) + (lane & 15)] = acc[r] + bb;
  }
}

// ==========================================================================
extern "C" void kernel_launch(void* const* d_in, const int* in_sizes, int n_in,
                              void* d_out, int out_size, void* d_ws, size_t ws_size,
                              hipStream_t stream) {
  (void)in_sizes; (void)n_in; (void)out_size; (void)ws_size;
  const float* trg   = (const float*)d_in[0];
  const float* w_ih  = (const float*)d_in[1];
  const float* w_hh  = (const float*)d_in[2];
  const float* b_ih  = (const float*)d_in[3];
  const float* b_hh  = (const float*)d_in[4];
  const float* w_out = (const float*)d_in[5];
  const float* b_out = (const float*)d_in[6];
  float* out = (float*)d_out;

  // workspace layout (bytes)
  uint8_t* ws = (uint8_t*)d_ws;
  uint16_t* wih_s  = (uint16_t*)(ws + 0);           //  2 MiB : 4H x I bf16 B-panels
  uint16_t* whh_s  = (uint16_t*)(ws + 2097152);     //  8 MiB : 4H x H bf16 B-panels
  uint16_t* wout_s = (uint16_t*)(ws + 10485760);    // .5 MiB : O x H  bf16 B-panels
  float*    bias4  = (float*)  (ws + 11010048);     // 16 KiB : b_ih + b_hh
  float*    cst    = (float*)  (ws + 11026432);     // .5 MiB : c state fp32 [B,H]
  uint16_t* hbuf   = (uint16_t*)(ws + 11550720);    // .5 MiB : h swizzled, double buffered
  uint16_t* xsw    = (uint16_t*)(ws + 12582912);    // 64 MiB : trg swizzled A-panels

  hipMemsetAsync(cst,  0, (size_t)BB * HH * sizeof(float), stream);
  hipMemsetAsync(hbuf, 0, (size_t)2 * BB * HH * sizeof(uint16_t), stream);

  swizzle_b_kernel<<<(4 * HH * II) / 256, 256, 0, stream>>>(w_ih,  wih_s,  II, 4 * HH * II);
  swizzle_b_kernel<<<(4 * HH * HH) / 256, 256, 0, stream>>>(w_hh,  whh_s,  HH, 4 * HH * HH);
  swizzle_b_kernel<<<(OO * HH)     / 256, 256, 0, stream>>>(w_out, wout_s, HH, OO * HH);
  swizzle_x_kernel<<<(BB * LL * II) / 256, 256, 0, stream>>>(trg, xsw, BB * LL * II);
  bias_kernel<<<(4 * HH) / 256, 256, 0, stream>>>(b_ih, b_hh, bias4, 4 * HH);

  const size_t hstride = (size_t)BB * HH;   // elements per h buffer
  for (int t = 0; t < LL; ++t) {
    int t_in = (t == 0) ? 0 : t - 1;        // teacher forcing: step t consumes trg[t-1]
    const uint16_t* h_in  = hbuf + (size_t)(t & 1) * hstride;
    uint16_t*       h_out = hbuf + (size_t)((t + 1) & 1) * hstride;
    lstm_step_kernel<<<dim3(HH / 16), dim3(256), 0, stream>>>(
        xsw, t_in, wih_s, whh_s, bias4, h_in, h_out, cst,
        wout_s, b_out, out, /*emit_t=*/t - 1);
  }
  // final step's h -> out[:, L-1, :]
  out_proj_kernel<<<dim3(OO / 16), dim3(256), 0, stream>>>(
      hbuf + (size_t)(LL & 1) * hstride, wout_s, b_out, out, LL - 1);
}
